// MultiHeadSelfAttention_58093727646383
// MI455X (gfx1250) — matrix operand order repaired and checked by reading, not compiled, not run on hardware
//
#include <hip/hip_runtime.h>
#include <hip/hip_fp16.h>

// Problem constants (match reference)
#define B_  4
#define T_  2048
#define C_  1024
#define H_  16
#define DH_ 64
#define C3_ 3072

typedef __attribute__((ext_vector_type(16))) _Float16 v16h;
typedef __attribute__((ext_vector_type(8)))  _Float16 v8h;
typedef __attribute__((ext_vector_type(8)))  float    v8f;

static __device__ __forceinline__ v8f wmma_f16(v16h a, v16h b, v8f c) {
  // D = A(16x32 f16) * B(32x16 f16) + C(16x16 f32)
  return __builtin_amdgcn_wmma_f32_16x16x32_f16(false, a, false, b, (short)0, c,
                                                false, false);
}

// A-operand (16x32, MxK) per-lane layout (contiguous k order, matches B):
//   lane l: row m = l%16, hi = l/16
//   elements e=0..15 -> k = k0 + hi*16 + e
static __device__ __forceinline__ v16h load_a_frag(const _Float16* row, int k0, int hi) {
  v16h a = *(const v16h*)(row + k0 + hi * 16);
  return a;
}

#define WAIT_ASYNC() asm volatile("s_wait_asynccnt 0x0" ::: "memory")
#define WAIT_DS()    asm volatile("s_wait_dscnt 0x0" ::: "memory")

// Async-stage one 128col x 32k f16 B-tile (8 KB) into LDS.
// LDS layout: [col][k] (64 B per column). 512 x 16B chunks, 4 per thread.
// Uses GVS addressing: mem = SADDR(u64) + VADDR(u32).
static __device__ __forceinline__ void stage_b(_Float16* dstLds, const _Float16* W,
                                               int oc0, int k0, int tid) {
  unsigned long long base = (unsigned long long)(const void*)(W + (size_t)oc0 * C_ + k0);
  unsigned ldsbase = (unsigned)(size_t)(void*)dstLds;  // low 32 bits = LDS byte addr
#pragma unroll
  for (int i = 0; i < 4; ++i) {
    int c = tid + i * 128;                      // chunk id in [0, 512)
    unsigned voff = (unsigned)((c >> 2) * (C_ * 2) + (c & 3) * 16);
    unsigned lds  = ldsbase + (unsigned)c * 16;
    asm volatile("global_load_async_to_lds_b128 %0, %1, %2"
                 :: "v"(lds), "v"(voff), "s"(base) : "memory");
  }
}

// Shared GEMM mainloop: C[64x128] tile, 4 waves x (16x128 per wave),
// B double-buffered through LDS via async copies, A from global.
static __device__ __forceinline__ void gemm_mainloop(const _Float16* __restrict__ X,
                                                     const _Float16* __restrict__ W,
                                                     _Float16* BsBase,  // shared, 2*128*32
                                                     int row0, int oc0,
                                                     int ln, int hi, int tid,
                                                     v8f acc[8]) {
  stage_b(BsBase, W, oc0, 0, tid);
  WAIT_ASYNC();
  __syncthreads();

  const _Float16* arow = X + (size_t)(row0 + ln) * C_;
  int buf = 0;
  for (int k0 = 0; k0 < C_; k0 += 32) {
    _Float16* cur = BsBase + buf * (128 * 32);
    if (k0 + 32 < C_)                                  // uniform branch
      stage_b(BsBase + (buf ^ 1) * (128 * 32), W, oc0, k0 + 32, tid);

    v16h a = load_a_frag(arow, k0, hi);
#pragma unroll
    for (int nt = 0; nt < 8; ++nt) {
      v16h b = *(const v16h*)(cur + (nt * 16 + ln) * 32 + hi * 16);  // k = hi*16+e
      acc[nt] = wmma_f16(a, b, acc[nt]);
    }

    WAIT_DS();      // this wave's LDS reads done before buffers are re-staged
    WAIT_ASYNC();   // this wave's async stage landed in LDS
    __syncthreads();
    buf ^= 1;
  }
}

// ---------------------------------------------------------------------------
// f32 -> f16 conversion
// ---------------------------------------------------------------------------
__global__ void cvt_f32_f16(const float* __restrict__ src, _Float16* __restrict__ dst, int n) {
  int i = blockIdx.x * blockDim.x + threadIdx.x;
  if (i < n) dst[i] = (_Float16)src[i];
}

// ---------------------------------------------------------------------------
// QKV GEMM: Y[row, oc] = X[row,:] . W[oc,:] + bias[oc]
// Epilogue scatters into q16/k16 (B,H,T,dh) and vT16 (B,H,dh,T), f16.
// ---------------------------------------------------------------------------
__global__ __launch_bounds__(128)
void gemm_qkv(const _Float16* __restrict__ X, const _Float16* __restrict__ W,
              const float* __restrict__ bias,
              _Float16* __restrict__ q16, _Float16* __restrict__ k16,
              _Float16* __restrict__ vT16) {
  __shared__ __align__(128) _Float16 Bs[2 * 128 * 32];
  const int tid  = threadIdx.x;
  const int lane = tid & 31;
  const int wave = tid >> 5;
  const int hi = lane >> 4;
  const int ln = lane & 15;
  const int row0 = blockIdx.x * 64 + wave * 16;   // [0, 8192)
  const int oc0  = blockIdx.y * 128;              // [0, 3072)

  v8f acc[8];
#pragma unroll
  for (int nt = 0; nt < 8; ++nt) {
    float bz = bias[oc0 + nt * 16 + ln];  // bias depends on column only
#pragma unroll
    for (int r = 0; r < 8; ++r) acc[nt][r] = bz;
  }

  gemm_mainloop(X, W, Bs, row0, oc0, ln, hi, tid, acc);

  const int bidx  = row0 >> 11;     // row0 / T_
  const int tbase = row0 & (T_ - 1);
#pragma unroll
  for (int nt = 0; nt < 8; ++nt) {
    int oc  = oc0 + nt * 16 + ln;
    int seg = oc >> 10;             // 0 = Q, 1 = K, 2 = V
    int ocl = oc & (C_ - 1);
    int hd  = ocl >> 6;
    int d   = ocl & (DH_ - 1);
#pragma unroll
    for (int r = 0; r < 8; ++r) {
      int t = tbase + r + 8 * hi;   // C-layout: M = r + 8*(lane/16)
      _Float16 val = (_Float16)acc[nt][r];
      size_t bh = (size_t)bidx * H_ + hd;
      if (seg == 0)      q16[(bh * T_ + t) * DH_ + d] = val;
      else if (seg == 1) k16[(bh * T_ + t) * DH_ + d] = val;
      else               vT16[(bh * DH_ + d) * T_ + t] = val;  // transposed V
    }
  }
}

// ---------------------------------------------------------------------------
// In-place RoPE on q16 and k16 (f32 math). One thread per (b,h,t,pair).
// ---------------------------------------------------------------------------
__global__ void rope_kernel(_Float16* __restrict__ q, _Float16* __restrict__ k,
                            const float* __restrict__ fcos, const float* __restrict__ fsin) {
  size_t idx = (size_t)blockIdx.x * blockDim.x + threadIdx.x;  // B*H*T*32
  if (idx >= (size_t)B_ * H_ * T_ * (DH_ / 2)) return;
  int i = (int)(idx & 31);
  size_t row = idx >> 5;            // (b*H + h)*T + t
  int t = (int)(row & (T_ - 1));
  float c = fcos[t * 32 + i];
  float s = fsin[t * 32 + i];
  size_t base = row * DH_ + (size_t)i * 2;
  float a0 = (float)q[base], b0 = (float)q[base + 1];
  q[base]     = (_Float16)(a0 * c - b0 * s);
  q[base + 1] = (_Float16)(a0 * s + b0 * c);
  a0 = (float)k[base]; b0 = (float)k[base + 1];
  k[base]     = (_Float16)(a0 * c - b0 * s);
  k[base + 1] = (_Float16)(a0 * s + b0 * c);
}

// ---------------------------------------------------------------------------
// Flash attention, causal. One wave (32 threads) per 16-query tile per (b,h).
// S = Q.K^T via WMMA; online softmax in f32 with wave32 cross-lane
// reductions; P staged through LDS (C-layout -> A-layout); O += P.V via WMMA
// on transposed V. Writes O as f16 in (B,T,C) layout.
// ---------------------------------------------------------------------------
__global__ __launch_bounds__(32)
void flash_attn(const _Float16* __restrict__ q16, const _Float16* __restrict__ k16,
                const _Float16* __restrict__ vT16, _Float16* __restrict__ o16) {
  __shared__ __align__(64) _Float16 P[16 * 32];

  const int lane = threadIdx.x;
  const int hi = lane >> 4;
  const int ln = lane & 15;
  const int q0 = blockIdx.x * 16;
  const int h  = blockIdx.y;
  const int b  = blockIdx.z;

  const _Float16* Q = q16 + ((size_t)b * H_ + h) * T_ * DH_;
  const _Float16* K = k16 + ((size_t)b * H_ + h) * T_ * DH_;
  const _Float16* V = vT16 + ((size_t)b * H_ + h) * DH_ * T_;

  // Q A-fragments for the two dh K-steps (dh = 64 = 2 * 32)
  const _Float16* qrow = Q + (size_t)(q0 + ln) * DH_;
  v16h qa0 = load_a_frag(qrow, 0, hi);
  v16h qa1 = load_a_frag(qrow, 32, hi);

  v8f o_acc[4];
#pragma unroll
  for (int nt = 0; nt < 4; ++nt)
#pragma unroll
    for (int r = 0; r < 8; ++r) o_acc[nt][r] = 0.0f;

  float mrow[8], lrow[8];
#pragma unroll
  for (int r = 0; r < 8; ++r) { mrow[r] = -__builtin_inff(); lrow[r] = 0.0f; }

  const float scale = 0.125f;       // 1/sqrt(64)
  const int kend = q0 + 16;         // keys [0, kend) are (partially) live

  for (int j0 = 0; j0 < kend; j0 += 32) {
    float pv[2][8];
    float mt[8];
#pragma unroll
    for (int r = 0; r < 8; ++r) mt[r] = -__builtin_inff();

#pragma unroll
    for (int sub = 0; sub < 2; ++sub) {
      const int kc = j0 + sub * 16 + ln;   // this lane's key column
      v8f s;
#pragma unroll
      for (int r = 0; r < 8; ++r) s[r] = 0.0f;
      if (j0 + sub * 16 < kend) {          // wave-uniform guard
        const _Float16* krow = K + (size_t)kc * DH_;
        v16h kb0 = *(const v16h*)(krow + hi * 16);       // dh 0..31
        v16h kb1 = *(const v16h*)(krow + 32 + hi * 16);  // dh 32..63
        s = wmma_f16(qa0, kb0, s);
        s = wmma_f16(qa1, kb1, s);
      }
#pragma unroll
      for (int r = 0; r < 8; ++r) {
        int qg = q0 + r + 8 * hi;
        float val = s[r] * scale;
        if (kc > qg) val = -__builtin_inff();   // causal mask
        pv[sub][r] = val;
        mt[r] = fmaxf(mt[r], val);
      }
    }

    // row-max across the 16 lanes sharing a row (wave32 halves)
#pragma unroll
    for (int r = 0; r < 8; ++r) {
      float v = mt[r];
      v = fmaxf(v, __shfl_xor(v, 1, 32));
      v = fmaxf(v, __shfl_xor(v, 2, 32));
      v = fmaxf(v, __shfl_xor(v, 4, 32));
      v = fmaxf(v, __shfl_xor(v, 8, 32));
      mt[r] = v;
    }

    float fac[8];
#pragma unroll
    for (int r = 0; r < 8; ++r) {
      float mnew = fmaxf(mrow[r], mt[r]);   // finite: every row has >=1 live key
      fac[r] = __expf(mrow[r] - mnew);
      mrow[r] = mnew;
    }

    float lad[8];
#pragma unroll
    for (int r = 0; r < 8; ++r) lad[r] = 0.0f;
#pragma unroll
    for (int sub = 0; sub < 2; ++sub)
#pragma unroll
      for (int r = 0; r < 8; ++r) {
        float p = __expf(pv[sub][r] - mrow[r]);
        lad[r] += p;
        P[(r + 8 * hi) * 32 + sub * 16 + ln] = (_Float16)p;  // row-major 16x32
      }

#pragma unroll
    for (int r = 0; r < 8; ++r) {
      float v = lad[r];
      v += __shfl_xor(v, 1, 32);
      v += __shfl_xor(v, 2, 32);
      v += __shfl_xor(v, 4, 32);
      v += __shfl_xor(v, 8, 32);
      lrow[r] = lrow[r] * fac[r] + v;
    }

#pragma unroll
    for (int nt = 0; nt < 4; ++nt)
#pragma unroll
      for (int r = 0; r < 8; ++r) o_acc[nt][r] *= fac[r];

    __syncthreads();  // order LDS P stores before A-fragment loads

    v16h pa = load_a_frag(&P[ln * 32], 0, hi);   // A = P (16x32 over keys)
#pragma unroll
    for (int nt = 0; nt < 4; ++nt) {
      const _Float16* vrow = V + (size_t)(nt * 16 + ln) * T_;   // vT row = dh idx
      v16h vb = *(const v16h*)(vrow + j0 + hi * 16);            // k = hi*16 + e
      o_acc[nt] = wmma_f16(pa, vb, o_acc[nt]);
    }

    __syncthreads();  // P reused next iteration
  }

  // normalize + store O in (B,T,C) f16
#pragma unroll
  for (int nt = 0; nt < 4; ++nt)
#pragma unroll
    for (int r = 0; r < 8; ++r) {
      int qm = r + 8 * hi;
      float val = o_acc[nt][r] / lrow[r];
      o16[((size_t)b * T_ + q0 + qm) * C_ + h * DH_ + nt * 16 + ln] = (_Float16)val;
    }
}

// ---------------------------------------------------------------------------
// Output projection: out[row, oc] = O[row,:] . Wp[oc,:] + bias[oc], f32 out.
// ---------------------------------------------------------------------------
__global__ __launch_bounds__(128)
void gemm_proj(const _Float16* __restrict__ X, const _Float16* __restrict__ W,
               const float* __restrict__ bias, float* __restrict__ out) {
  __shared__ __align__(128) _Float16 Bs[2 * 128 * 32];
  const int tid  = threadIdx.x;
  const int lane = tid & 31;
  const int wave = tid >> 5;
  const int hi = lane >> 4;
  const int ln = lane & 15;
  const int row0 = blockIdx.x * 64 + wave * 16;   // [0, 8192)
  const int oc0  = blockIdx.y * 128;              // [0, 1024)

  v8f acc[8];
#pragma unroll
  for (int nt = 0; nt < 8; ++nt) {
    float bz = bias[oc0 + nt * 16 + ln];
#pragma unroll
    for (int r = 0; r < 8; ++r) acc[nt][r] = bz;
  }

  gemm_mainloop(X, W, Bs, row0, oc0, ln, hi, tid, acc);

#pragma unroll
  for (int nt = 0; nt < 8; ++nt)
#pragma unroll
    for (int r = 0; r < 8; ++r)
      out[(size_t)(row0 + r + 8 * hi) * C_ + oc0 + nt * 16 + ln] = acc[nt][r];
}

// ---------------------------------------------------------------------------
// Launch
// ---------------------------------------------------------------------------
extern "C" void kernel_launch(void* const* d_in, const int* in_sizes, int n_in,
                              void* d_out, int out_size, void* d_ws, size_t ws_size,
                              hipStream_t stream) {
  (void)in_sizes; (void)n_in; (void)out_size; (void)ws_size;

  const float* x      = (const float*)d_in[0];
  // d_in[1] = mask (causal, handled analytically)
  const float* fcos   = (const float*)d_in[2];
  const float* fsin   = (const float*)d_in[3];
  const float* qkv_w  = (const float*)d_in[4];
  const float* qkv_b  = (const float*)d_in[5];
  const float* proj_w = (const float*)d_in[6];
  const float* proj_b = (const float*)d_in[7];
  float* out = (float*)d_out;

  // Workspace layout (all 256B-aligned, ~88 MB total)
  char* ws = (char*)d_ws;
  size_t off = 0;
  _Float16* x16  = (_Float16*)(ws + off); off += (size_t)B_ * T_ * C_ * 2;
  _Float16* wq16 = (_Float16*)(ws + off); off += (size_t)C3_ * C_ * 2;
  _Float16* wp16 = (_Float16*)(ws + off); off += (size_t)C_ * C_ * 2;
  _Float16* q16  = (_Float16*)(ws + off); off += (size_t)B_ * H_ * T_ * DH_ * 2;
  _Float16* k16  = (_Float16*)(ws + off); off += (size_t)B_ * H_ * T_ * DH_ * 2;
  _Float16* vT16 = (_Float16*)(ws + off); off += (size_t)B_ * H_ * T_ * DH_ * 2;
  _Float16* o16  = (_Float16*)(ws + off); off += (size_t)B_ * T_ * C_ * 2;

  const int nx = B_ * T_ * C_;       // 8388608
  const int nwq = C3_ * C_;          // 3145728
  const int nwp = C_ * C_;           // 1048576

  cvt_f32_f16<<<(nx + 255) / 256, 256, 0, stream>>>(x, x16, nx);
  cvt_f32_f16<<<(nwq + 255) / 256, 256, 0, stream>>>(qkv_w, wq16, nwq);
  cvt_f32_f16<<<(nwp + 255) / 256, 256, 0, stream>>>(proj_w, wp16, nwp);

  gemm_qkv<<<dim3((B_ * T_) / 64, C3_ / 128, 1), 128, 0, stream>>>(
      x16, wq16, qkv_b, q16, k16, vT16);

  const int nrope = B_ * H_ * T_ * (DH_ / 2);   // 4194304
  rope_kernel<<<(nrope + 255) / 256, 256, 0, stream>>>(q16, k16, fcos, fsin);

  flash_attn<<<dim3(T_ / 16, H_, B_), 32, 0, stream>>>(q16, k16, vT16, o16);

  gemm_proj<<<dim3((B_ * T_) / 64, C_ / 128, 1), 128, 0, stream>>>(
      o16, wp16, proj_b, out);
}
